// MoE_26731876450392
// MI455X (gfx1250) — matrix operand order repaired and checked by reading, not compiled, not run on hardware
//
#include <hip/hip_runtime.h>
#include <hip/hip_bf16.h>

typedef __attribute__((ext_vector_type(16))) _Float16 v16h;
typedef __attribute__((ext_vector_type(8)))  _Float16 v8h;
typedef __attribute__((ext_vector_type(4)))  _Float16 v4h;
typedef __attribute__((ext_vector_type(8)))  float    v8f;

#define NEXP   8
#define DMODEL 1024
#define DFFN   4096
#define SEQ    2048
#define NBATCH 4
#define TOPK   512

// ---------------------------------------------------------------------------
// 1) Router: logits = x . choice^T, softmax over experts -> probs[b,s,8]
//    One wave (32 lanes) per token; 8 accumulators per lane; xor-reduce.
// ---------------------------------------------------------------------------
__global__ __launch_bounds__(256)
void router_kernel(const float* __restrict__ x, const float* __restrict__ choice,
                   float* __restrict__ probs)
{
    const int tid   = threadIdx.x;
    const int wave  = tid >> 5;
    const int lane  = tid & 31;
    const int token = blockIdx.x * 8 + wave;            // [0, NBATCH*SEQ)
    const float* xrow = x + (size_t)token * DMODEL;

    float acc[NEXP];
#pragma unroll
    for (int e = 0; e < NEXP; ++e) acc[e] = 0.f;

    for (int j = 0; j < DMODEL / 32; ++j) {
        const int d = lane + j * 32;
        const float xv = xrow[d];
#pragma unroll
        for (int e = 0; e < NEXP; ++e) acc[e] += xv * choice[e * DMODEL + d];
    }
#pragma unroll
    for (int e = 0; e < NEXP; ++e) {
#pragma unroll
        for (int off = 16; off > 0; off >>= 1)
            acc[e] += __shfl_xor(acc[e], off, 32);
    }
    if (lane == 0) {
        float mx = acc[0];
#pragma unroll
        for (int e = 1; e < NEXP; ++e) mx = fmaxf(mx, acc[e]);
        float p[NEXP], sum = 0.f;
#pragma unroll
        for (int e = 0; e < NEXP; ++e) { p[e] = __expf(acc[e] - mx); sum += p[e]; }
        const float inv = 1.f / sum;
#pragma unroll
        for (int e = 0; e < NEXP; ++e) probs[(size_t)token * NEXP + e] = p[e] * inv;
    }
}

// ---------------------------------------------------------------------------
// 2) Top-512 per (b,e): bitonic sort of 2048 (prob, idx) pairs in LDS.
//    Order: descending prob, ascending idx on ties (matches jax top_k set).
// ---------------------------------------------------------------------------
__global__ __launch_bounds__(256)
void topk_kernel(const float* __restrict__ probs, int* __restrict__ I,
                 float* __restrict__ G, int* __restrict__ slot,
                 float* __restrict__ gate)
{
    __shared__ float skey[SEQ];
    __shared__ int   sidx[SEQ];
    const int be = blockIdx.x;                 // b*8+e
    const int b  = be >> 3, e = be & 7;
    const int tid = threadIdx.x;

    for (int i = tid; i < SEQ; i += 256) {
        skey[i] = probs[((size_t)b * SEQ + i) * NEXP + e];
        sidx[i] = i;
    }
    __syncthreads();

    for (int ksz = 2; ksz <= SEQ; ksz <<= 1) {
        for (int j = ksz >> 1; j > 0; j >>= 1) {
            for (int i = tid; i < SEQ; i += 256) {
                const int ixj = i ^ j;
                if (ixj > i) {
                    const bool up = ((i & ksz) == 0);
                    const float ka = skey[i], kb = skey[ixj];
                    const int   ia = sidx[i], ib = sidx[ixj];
                    // "l": element i belongs before element ixj in final order
                    const bool l = (ka > kb) || (ka == kb && ia < ib);
                    if (up != l) {
                        skey[i] = kb; skey[ixj] = ka;
                        sidx[i] = ib; sidx[ixj] = ia;
                    }
                }
            }
            __syncthreads();
        }
    }

    for (int i = tid; i < SEQ; i += 256) {
        slot[(size_t)be * SEQ + i] = -1;
        gate[(size_t)be * SEQ + i] = 0.f;
    }
    __syncthreads();
    for (int k = tid; k < TOPK; k += 256) {
        const int   t = sidx[k];
        const float g = skey[k];
        I[(size_t)be * TOPK + k] = t;
        G[(size_t)be * TOPK + k] = g;
        slot[(size_t)be * SEQ + t] = k;
        gate[(size_t)be * SEQ + t] = g;
    }
}

// ---------------------------------------------------------------------------
// 3) Gather selected tokens, fp32 -> f16: x_in[be,512,1024]
// ---------------------------------------------------------------------------
__global__ __launch_bounds__(256)
void gather_kernel(const float* __restrict__ x, const int* __restrict__ I,
                   _Float16* __restrict__ xin)
{
    const int row = blockIdx.x;                 // be*512 + k
    const int be  = row >> 9;
    const int b   = be >> 3;
    const int tok = I[row];
    const float4 v = *(const float4*)(x + ((size_t)b * SEQ + tok) * DMODEL + threadIdx.x * 4);
    v4h h;
    h[0] = (_Float16)v.x; h[1] = (_Float16)v.y;
    h[2] = (_Float16)v.z; h[3] = (_Float16)v.w;
    *(v4h*)(xin + (size_t)row * DMODEL + threadIdx.x * 4) = h;
}

// ---------------------------------------------------------------------------
// 4/5) Tiled WMMA GEMM:  C[512, N] = act(A[512,K] . W[N,K]^T)   per (b,e)
//   A: f16 row-major (K contiguous).  W: fp32 row-major (K contiguous),
//   converted to f16 while staging to LDS.  Block tile 128x128x32, 8 waves,
//   each wave owns 64x32 = 4x2 v_wmma_f32_16x16x32_f16 accumulators.
//   Double-buffered LDS (ping-pong, one barrier per K-step): global loads for
//   tile n+1 are issued before the barrier so HBM latency runs under the
//   WMMAs of tile n.
// ---------------------------------------------------------------------------
#define BM 128
#define BN 128
#define BK 32
#define LDST 40   // padded LDS row stride in halfs (80B)

template<bool ACT>
__global__ __launch_bounds__(256)
void moe_gemm(const _Float16* __restrict__ Aall, const float* __restrict__ Wall,
              _Float16* __restrict__ outH, float* __restrict__ outF,
              const int Kdim, const int Ndim)
{
    __shared__ _Float16 As[2][BM * LDST];
    __shared__ _Float16 Bs[2][BN * LDST];

    const int be = blockIdx.z;                  // b*8+e
    const int e  = be & 7;
    const int m0 = blockIdx.y * BM;
    const int n0 = blockIdx.x * BN;

    const _Float16* A = Aall + (size_t)be * TOPK * Kdim;
    const float*    W = Wall + (size_t)e * Ndim * Kdim;

    const int tid  = threadIdx.x;
    const int lane = tid & 31;
    const int wave = tid >> 5;
    const int wm   = wave >> 2;                 // 0..1  (64 rows each)
    const int wn   = wave & 3;                  // 0..3  (32 cols each)
    const int hi   = lane >> 4;                 // lane half
    const int lr   = lane & 15;

    // per-thread staging coordinates (fixed across K-steps)
    const int arow0 = (tid      ) >> 2, ako0 = ((tid      ) & 3) * 8;
    const int arow1 = (tid + 256) >> 2, ako1 = ((tid + 256) & 3) * 8;
    int brow[4], bko[4];
#pragma unroll
    for (int c = 0; c < 4; ++c) {
        const int chunk = tid + c * 256;
        brow[c] = chunk >> 3;
        bko[c]  = (chunk & 7) * 4;
    }

    v8f acc[4][2] = {};

    v8h   areg[2];
    float4 wreg[4];

    // prefetch tile 0 into registers, stage into buffer 0
    areg[0] = *(const v8h*)(A + (size_t)(m0 + arow0) * Kdim + ako0);
    areg[1] = *(const v8h*)(A + (size_t)(m0 + arow1) * Kdim + ako1);
#pragma unroll
    for (int c = 0; c < 4; ++c)
        wreg[c] = *(const float4*)(W + (size_t)(n0 + brow[c]) * Kdim + bko[c]);

    *(v8h*)(&As[0][arow0 * LDST + ako0]) = areg[0];
    *(v8h*)(&As[0][arow1 * LDST + ako1]) = areg[1];
#pragma unroll
    for (int c = 0; c < 4; ++c) {
        v4h h4;
        h4[0] = (_Float16)wreg[c].x; h4[1] = (_Float16)wreg[c].y;
        h4[2] = (_Float16)wreg[c].z; h4[3] = (_Float16)wreg[c].w;
        *(v4h*)(&Bs[0][brow[c] * LDST + bko[c]]) = h4;
    }

    int cur = 0;
    for (int k0 = 0; k0 < Kdim; k0 += BK) {
        const bool has_next = (k0 + BK) < Kdim;
        // issue next tile's global loads before the barrier (latency overlap)
        if (has_next) {
            const int kn = k0 + BK;
            areg[0] = *(const v8h*)(A + (size_t)(m0 + arow0) * Kdim + kn + ako0);
            areg[1] = *(const v8h*)(A + (size_t)(m0 + arow1) * Kdim + kn + ako1);
#pragma unroll
            for (int c = 0; c < 4; ++c)
                wreg[c] = *(const float4*)(W + (size_t)(n0 + brow[c]) * Kdim + kn + bko[c]);
        }
        __syncthreads();   // publishes buf[cur]; retires prior reads of buf[cur^1]

        // A frag (16x32, lane m=lr): halfs [hi*8 .. +7] and [16+hi*8 .. +7]
        v16h afrag[4], bfrag[2];
#pragma unroll
        for (int mt = 0; mt < 4; ++mt) {
            const _Float16* ar = &As[cur][(wm * 64 + mt * 16 + lr) * LDST];
            const v8h lo = *(const v8h*)(ar + hi * 8);
            const v8h hv = *(const v8h*)(ar + 16 + hi * 8);
            afrag[mt] = __builtin_shufflevector(lo, hv,
                0,1,2,3,4,5,6,7,8,9,10,11,12,13,14,15);
        }
        // B frag (32x16, lane n=lr): W-row n, halfs [hi*8 .. +7] and [16+hi*8 .. +7]
#pragma unroll
        for (int nt = 0; nt < 2; ++nt) {
            const _Float16* br = &Bs[cur][(wn * 32 + nt * 16 + lr) * LDST];
            const v8h lo = *(const v8h*)(br + hi * 8);
            const v8h hv = *(const v8h*)(br + 16 + hi * 8);
            bfrag[nt] = __builtin_shufflevector(lo, hv,
                0,1,2,3,4,5,6,7,8,9,10,11,12,13,14,15);
        }
#pragma unroll
        for (int mt = 0; mt < 4; ++mt)
#pragma unroll
            for (int nt = 0; nt < 2; ++nt)
                acc[mt][nt] = __builtin_amdgcn_wmma_f32_16x16x32_f16(
                    false, afrag[mt], false, bfrag[nt],
                    (short)0, acc[mt][nt], false, false);

        // stage next tile into the other buffer (no barrier needed here)
        if (has_next) {
            const int nxt = cur ^ 1;
            *(v8h*)(&As[nxt][arow0 * LDST + ako0]) = areg[0];
            *(v8h*)(&As[nxt][arow1 * LDST + ako1]) = areg[1];
#pragma unroll
            for (int c = 0; c < 4; ++c) {
                v4h h4;
                h4[0] = (_Float16)wreg[c].x; h4[1] = (_Float16)wreg[c].y;
                h4[2] = (_Float16)wreg[c].z; h4[3] = (_Float16)wreg[c].w;
                *(v4h*)(&Bs[nxt][brow[c] * LDST + bko[c]]) = h4;
            }
        }
        cur ^= 1;
    }

    // Epilogue: C/D layout -> lane holds rows m_base+(hi*8)+r, col = lr
    const size_t outbase = (size_t)be * TOPK * Ndim;
#pragma unroll
    for (int mt = 0; mt < 4; ++mt) {
        const int mrow0 = m0 + wm * 64 + mt * 16 + hi * 8;
#pragma unroll
        for (int nt = 0; nt < 2; ++nt) {
            const int col = n0 + wn * 32 + nt * 16 + lr;
#pragma unroll
            for (int r = 0; r < 8; ++r) {
                const float v = acc[mt][nt][r];
                const size_t o = outbase + (size_t)(mrow0 + r) * Ndim + col;
                if constexpr (ACT) {
                    // silu via v_rcp_f32 (~1 ulp), avoids IEEE div chain
                    const float s = v * __builtin_amdgcn_rcpf(1.f + __expf(-v));
                    outH[o] = (_Float16)s;
                } else {
                    outF[o] = v;
                }
            }
        }
    }
}

// ---------------------------------------------------------------------------
// 6) Combine: x_out[b,s,:] = sum_e gate[b,e,s] * x_mlp[b,e,slot[b,e,s],:]
// ---------------------------------------------------------------------------
__global__ __launch_bounds__(256)
void combine_kernel(const float* __restrict__ xmlp, const int* __restrict__ slot,
                    const float* __restrict__ gate, float* __restrict__ out)
{
    const int bs = blockIdx.x;               // b*SEQ + s
    const int b  = bs >> 11;
    const int s  = bs & (SEQ - 1);
    const int d  = threadIdx.x * 4;
    float a0 = 0.f, a1 = 0.f, a2 = 0.f, a3 = 0.f;
#pragma unroll
    for (int e = 0; e < NEXP; ++e) {
        const size_t bes = (size_t)(b * NEXP + e) * SEQ + s;
        const int   sl = slot[bes];
        const float g  = gate[bes];
        if (sl >= 0) {
            const float4 v = *(const float4*)(
                xmlp + ((size_t)(b * NEXP + e) * TOPK + sl) * DMODEL + d);
            a0 += g * v.x; a1 += g * v.y; a2 += g * v.z; a3 += g * v.w;
        }
    }
    float4 o; o.x = a0; o.y = a1; o.z = a2; o.w = a3;
    *(float4*)(out + (size_t)bs * DMODEL + d) = o;
}

// ---------------------------------------------------------------------------
extern "C" void kernel_launch(void* const* d_in, const int* in_sizes, int n_in,
                              void* d_out, int out_size, void* d_ws, size_t ws_size,
                              hipStream_t stream)
{
    (void)in_sizes; (void)n_in; (void)out_size; (void)ws_size;
    const float* x      = (const float*)d_in[0];   // [4,2048,1024]
    const float* choice = (const float*)d_in[1];   // [8,1024]
    const float* w1     = (const float*)d_in[2];   // [8,4096,1024]
    const float* w2     = (const float*)d_in[3];   // [8,1024,4096]
    float* out = (float*)d_out;                    // [4,2048,1024]

    char* ws = (char*)d_ws;
    // workspace layout (256B-aligned regions), total ~236 MB
    float*    probs = (float*)(ws + 0);                       // 256 KB
    int*      I     = (int*)  (ws + 262144);                  // 64 KB
    float*    G     = (float*)(ws + 327680);                  // 64 KB
    int*      slot  = (int*)  (ws + 393216);                  // 256 KB
    float*    gate  = (float*)(ws + 655360);                  // 256 KB
    _Float16* xin   = (_Float16*)(ws + 917504);               // 32 MB
    _Float16* h     = (_Float16*)(ws + 917504 + 33554432);    // 128 MB
    float*    xmlp  = (float*)(ws + 917504 + 33554432 + 134217728); // 64 MB

    router_kernel<<<NBATCH * SEQ / 8, 256, 0, stream>>>(x, choice, probs);
    topk_kernel<<<NBATCH * NEXP, 256, 0, stream>>>(probs, I, G, slot, gate);
    gather_kernel<<<NBATCH * NEXP * TOPK, 256, 0, stream>>>(x, I, xin);
    moe_gemm<true ><<<dim3(DFFN / BN,   TOPK / BM, NBATCH * NEXP), 256, 0, stream>>>(
        xin, w1, h, (float*)nullptr, DMODEL, DFFN);
    moe_gemm<false><<<dim3(DMODEL / BN, TOPK / BM, NBATCH * NEXP), 256, 0, stream>>>(
        h, w2, (_Float16*)nullptr, xmlp, DFFN, DMODEL);
    combine_kernel<<<NBATCH * SEQ, 256, 0, stream>>>(xmlp, slot, gate, out);
}
